// AGRNN_50474455663044
// MI455X (gfx1250) — matrix operand order repaired and checked by reading, not compiled, not run on hardware
//
#include <hip/hip_runtime.h>
#include <hip/hip_bf16.h>

// ---------------------------------------------------------------------------
// Problem constants (from reference): B=256, P=20, H=4
//   N = 5120 nodes, E = 97280 edges, Er = 19456 readout edges
//   F = 1024, L = 300, S = 16, HD = 1024, C = 117, D_in = 2664
// ---------------------------------------------------------------------------
#define NNODES   5120
#define NEDGES   97280
#define NREAD    19456
#define FDIM     1024
#define LDIM     300
#define SDIM     16
#define HDDIM    1024
#define CDIM     117
#define DIN      2664   // 1024 + 300 + 16 + 300 + 1024

typedef __attribute__((ext_vector_type(16))) __bf16 v16bf;
typedef __attribute__((ext_vector_type(8)))  __bf16 v8bf;
typedef __attribute__((ext_vector_type(8)))  float  v8f;

__device__ __forceinline__ unsigned short f2bf(float f) {
    unsigned int u = __float_as_uint(f);
    u += 0x7FFFu + ((u >> 16) & 1u);          // round-to-nearest-even
    return (unsigned short)(u >> 16);
}

// ---------------------------------------------------------------------------
// fp32 -> bf16 (row-major copy)
// ---------------------------------------------------------------------------
__global__ void cvt_f32_bf16(const float* __restrict__ s,
                             unsigned short* __restrict__ d, int n) {
    int i = blockIdx.x * blockDim.x + threadIdx.x;
    if (i < n) d[i] = f2bf(s[i]);
}

// fp32 [K][N] -> bf16 [N][K] (transpose; weights only, done once)
__global__ void cvt_transpose_bf16(const float* __restrict__ src,
                                   unsigned short* __restrict__ dst,
                                   int K, int N) {
    int i = blockIdx.x * blockDim.x + threadIdx.x;
    if (i < K * N) {
        int k = i / N, n = i - k * N;
        dst[(size_t)n * K + k] = f2bf(src[i]);
    }
}

// spb[r][0:16] = bf16(spatial_feat[readout_eid[r]][0:16])
__global__ void gather_spatial_bf16(const float* __restrict__ spat,
                                    const int* __restrict__ reid,
                                    unsigned short* __restrict__ spb, int n) {
    int i = blockIdx.x * blockDim.x + threadIdx.x;
    if (i < n) {
        int r = i >> 4, c = i & 15;
        spb[i] = f2bf(spat[(size_t)reid[r] * 16 + c]);
    }
}

// tab[r] = {src_node, dst_node}
__global__ void build_tab(const int* __restrict__ esrc, const int* __restrict__ edst,
                          const int* __restrict__ reid, int* __restrict__ tab, int n) {
    int i = blockIdx.x * blockDim.x + threadIdx.x;
    if (i < n) {
        int e = reid[i];
        tab[2 * i + 0] = esrc[e];
        tab[2 * i + 1] = edst[e];
    }
}

// ---------------------------------------------------------------------------
// per-row attention scores: es[r] = h[r]·a[0:d], ed[r] = h[r]·a[d:2d]
// one wave per row (wave32)
// ---------------------------------------------------------------------------
__global__ __launch_bounds__(256)
void row_scores(const float* __restrict__ h, const float* __restrict__ a,
                float* __restrict__ es, float* __restrict__ ed, int n, int d) {
    int wave = threadIdx.x >> 5, lane = threadIdx.x & 31;
    int row = blockIdx.x * 8 + wave;
    if (row >= n) return;                      // uniform per wave
    float s0 = 0.f, s1 = 0.f;
    for (int c = lane; c < d; c += 32) {
        float v = h[(size_t)row * d + c];
        s0 += v * a[c];
        s1 += v * a[d + c];
    }
    for (int o = 16; o > 0; o >>= 1) {
        s0 += __shfl_down(s0, o, 32);
        s1 += __shfl_down(s1, o, 32);
    }
    if (lane == 0) { es[row] = s0; ed[row] = s1; }
}

// ---------------------------------------------------------------------------
// GAT aggregation, structured graph: each dst node has exactly 19 in-edges
// (all other boxes in the same image). Exact segment-softmax semantics.
// out = bf16(relu(x + sum_q alpha_q * h[src_q]))
// ---------------------------------------------------------------------------
__global__ __launch_bounds__(128)
void gat_aggregate(const float* __restrict__ x, const float* __restrict__ h,
                   const float* __restrict__ es, const float* __restrict__ ed,
                   unsigned short* __restrict__ out, int d) {
    int node = blockIdx.x;
    int p = node % 20, base = node - p;
    __shared__ float s_a[19];
    int tid = threadIdx.x;
    if (tid == 0) {
        float e[19];
        float edst = ed[node];
        float emax = -3.4e38f;
        for (int q = 0; q < 19; ++q) {
            int src = base + q + (q >= p);
            float v = es[src] + edst;
            v = v > 0.f ? v : 0.2f * v;        // leaky_relu(0.2)
            e[q] = v;
            emax = fmaxf(emax, v);
        }
        float den = 0.f;
        for (int q = 0; q < 19; ++q) { e[q] = expf(e[q] - emax); den += e[q]; }
        float inv = 1.f / (den + 1e-9f);
        for (int q = 0; q < 19; ++q) s_a[q] = e[q] * inv;
    }
    __syncthreads();
    for (int c = tid; c < d; c += 128) {
        float m = 0.f;
        for (int q = 0; q < 19; ++q) {
            int src = base + q + (q >= p);
            m += s_a[q] * h[(size_t)src * d + c];
        }
        float v = x[(size_t)node * d + c] + m;
        out[(size_t)node * d + c] = f2bf(fmaxf(v, 0.f));
    }
}

// ---------------------------------------------------------------------------
// A-tile chunk loaders: return 4 packed bf16 (k multiple of 4), branchless
// (clamped address + cndmask select instead of exec-mask branches)
// ---------------------------------------------------------------------------
struct DirectLoader {
    const unsigned short* A;
    int M, K, lda;
    __device__ __forceinline__ uint2 chunk(int r, int k) const {
        bool ok = (r < M) && (k < K);          // K % 4 == 0: chunk all-in or all-out
        size_t idx = ok ? ((size_t)r * lda + k) : 0;
        uint2 v = *(const uint2*)(A + idx);
        v.x = ok ? v.x : 0u;
        v.y = ok ? v.y : 0u;
        return v;
    }
};

// Fused concat gather for the predictor row r:
//   [hv_new[d](1024) | hl_new[d](300) | spb[r](16) | hl_new[s](300) | hv_new[s](1024)]
// All region boundaries (1024,1324,1340,1640,2664) are multiples of 4 ->
// 4-element chunks never straddle; pointer chosen via nested selects (no branches).
struct GatherLoader {
    const int* tab;             // [NREAD][2] = {src, dst}
    const unsigned short* hv;   // [NNODES x 1024] bf16
    const unsigned short* hl;   // [NNODES x 300]  bf16
    const unsigned short* spb;  // [NREAD x 16]    bf16 (pre-gathered spatial)
    int M;
    __device__ __forceinline__ uint2 chunk(int r, int k) const {
        bool rok = r < M;
        int rr = rok ? r : 0;
        int s = tab[2 * rr + 0], d = tab[2 * rr + 1];
        const unsigned short* p =
            (k < 1324) ? ((k < 1024) ? hv + (size_t)d * 1024 + k
                                     : hl + (size_t)d * 300 + (k - 1024))
          : (k < 1640) ? ((k < 1340) ? spb + (size_t)rr * 16 + (k - 1324)
                                     : hl + (size_t)s * 300 + (k - 1340))
                       : hv + (size_t)s * 1024 + (k < DIN ? (k - 1640) : 0);
        uint2 v = *(const uint2*)p;
        bool ok = rok && (k < DIN);
        v.x = ok ? v.x : 0u;
        v.y = ok ? v.y : 0u;
        return v;
    }
};

// ---------------------------------------------------------------------------
// Tiled bf16 WMMA GEMM:  C[M,N] = A[M,K] @ B[K,N], B passed pre-transposed as
// Bt[N][K].  Block tile 128x128, BK=32, 256 threads = 8 waves (2x4 grid),
// each wave -> 64x32 of C via 4x2 v_wmma_f32_16x16x32_bf16 fragments.
// MODE 0: C fp32    MODE 1: relu(C + bias) -> bf16    MODE 2: C + bias -> fp32
// ---------------------------------------------------------------------------
template <typename Loader, int MODE>
__global__ __launch_bounds__(256)
void gemm_wmma(Loader ld, const unsigned short* __restrict__ Bt,
               int M, int N, int K, const float* __restrict__ bias,
               float* __restrict__ Cf, unsigned short* __restrict__ Cbf, int ldc) {
    __shared__ unsigned short sA[128 * 40];   // [row][k], stride 40 (bank-conflict pad)
    __shared__ unsigned short sB[128 * 40];   // [n][k]  (K-contiguous), stride 40

    const int tid  = threadIdx.x;
    const int bm   = blockIdx.y * 128, bn = blockIdx.x * 128;
    const int wave = tid >> 5, lane = tid & 31;
    const int half = lane >> 4, l16 = lane & 15;
    const int wm   = wave >> 2;               // 0..1  (64 rows each)
    const int wn   = wave & 3;                // 0..3  (32 cols each)

    v8f acc[4][2];
    for (int mf = 0; mf < 4; ++mf)
        for (int nf = 0; nf < 2; ++nf)
            for (int j = 0; j < 8; ++j) acc[mf][nf][j] = 0.f;

    const int kt = (K + 31) >> 5;
    for (int t = 0; t < kt; ++t) {
        const int k0 = t << 5;
        // stage A: 128x32 bf16 = 1024 b64 chunks, 4 per thread
        #pragma unroll
        for (int i = 0; i < 4; ++i) {
            int c = tid + (i << 8);                   // 0..1023
            int r = c >> 3, co = (c & 7) << 2;        // row, k-offset (mult of 4)
            uint2 v = ld.chunk(bm + r, k0 + co);
            *(uint2*)&sA[r * 40 + co] = v;
        }
        // stage B from Bt[N][K]: contiguous along K, same pattern
        #pragma unroll
        for (int i = 0; i < 4; ++i) {
            int c = tid + (i << 8);
            int n = c >> 3, co = (c & 7) << 2;
            bool ok = (bn + n < N) && (k0 + co < K);
            size_t idx = ok ? ((size_t)(bn + n) * K + k0 + co) : 0;
            uint2 v = *(const uint2*)(Bt + idx);
            v.x = ok ? v.x : 0u;
            v.y = ok ? v.y : 0u;
            *(uint2*)&sB[n * 40 + co] = v;
        }
        // speculative prefetch of next K-tile of Bt (global_prefetch_b8)
        if (t + 1 < kt) {
            int n = tid >> 1, co = (tid & 1) << 4;
            bool ok = (bn + n < N) && (k0 + 32 + co < K);
            size_t idx = ok ? ((size_t)(bn + n) * K + k0 + 32 + co) : 0;
            __builtin_prefetch(Bt + idx, 0, 1);
        }
        __syncthreads();

        // A fragment (16x32 bf16, ISA layout): lane half 0 -> K {0..7,16..23},
        // half 1 -> K {8..15,24..31}; row = l16.
        v16bf af[4], bfr[2];
        #pragma unroll
        for (int mf = 0; mf < 4; ++mf) {
            int ar = wm * 64 + mf * 16 + l16;
            v8bf lo = *(const v8bf*)&sA[ar * 40 + half * 8];
            v8bf hi = *(const v8bf*)&sA[ar * 40 + 16 + half * 8];
            #pragma unroll
            for (int i = 0; i < 8; ++i) { af[mf][i] = lo[i]; af[mf][i + 8] = hi[i]; }
        }
        // B fragment (32x16 bf16): col = l16, lane half 0 -> K {0..7,16..23},
        // half 1 -> K {8..15,24..31} (same order as A).
        #pragma unroll
        for (int nf = 0; nf < 2; ++nf) {
            int br = wn * 32 + nf * 16 + l16;
            v8bf lo = *(const v8bf*)&sB[br * 40 + half * 8];
            v8bf hi = *(const v8bf*)&sB[br * 40 + 16 + half * 8];
            #pragma unroll
            for (int i = 0; i < 8; ++i) { bfr[nf][i] = lo[i]; bfr[nf][i + 8] = hi[i]; }
        }
        #pragma unroll
        for (int mf = 0; mf < 4; ++mf)
            #pragma unroll
            for (int nf = 0; nf < 2; ++nf)
                acc[mf][nf] = __builtin_amdgcn_wmma_f32_16x16x32_bf16(
                    false, af[mf], false, bfr[nf], (short)0, acc[mf][nf], false, false);
        __syncthreads();
    }

    // epilogue: C layout (16x16 f32): VGPR j -> row half*8+j, col = l16
    #pragma unroll
    for (int mf = 0; mf < 4; ++mf)
        #pragma unroll
        for (int nf = 0; nf < 2; ++nf)
            #pragma unroll
            for (int j = 0; j < 8; ++j) {
                int row = bm + wm * 64 + mf * 16 + half * 8 + j;
                int col = bn + wn * 32 + nf * 16 + l16;
                if (row < M && col < N) {
                    float v = acc[mf][nf][j];
                    if (MODE >= 1) v += bias[col];
                    if (MODE == 1) {
                        Cbf[(size_t)row * ldc + col] = f2bf(fmaxf(v, 0.f));
                    } else {
                        Cf[(size_t)row * ldc + col] = v;
                    }
                }
            }
}

// ---------------------------------------------------------------------------
// host-side orchestration
// ---------------------------------------------------------------------------
extern "C" void kernel_launch(void* const* d_in, const int* in_sizes, int n_in,
                              void* d_out, int out_size, void* d_ws, size_t ws_size,
                              hipStream_t stream) {
    (void)in_sizes; (void)n_in; (void)out_size; (void)ws_size;

    const float* feat   = (const float*)d_in[0];
    const float* w2v    = (const float*)d_in[1];
    const float* spat   = (const float*)d_in[2];
    const float* W_v    = (const float*)d_in[3];
    const float* a_v    = (const float*)d_in[4];
    const float* W_l    = (const float*)d_in[5];
    const float* a_l    = (const float*)d_in[6];
    const float* W1     = (const float*)d_in[7];
    const float* b1     = (const float*)d_in[8];
    const float* W2     = (const float*)d_in[9];
    const float* b2     = (const float*)d_in[10];
    const int*   esrc   = (const int*)d_in[11];
    const int*   edst   = (const int*)d_in[12];
    const int*   reid   = (const int*)d_in[13];
    float* out = (float*)d_out;

    // workspace carve-up (~100 MB)
    char* w = (char*)d_ws;
    size_t off = 0;
    auto alloc = [&](size_t bytes) -> void* {
        void* p = w + off;
        off += (bytes + 255) & ~(size_t)255;
        return p;
    };
    unsigned short* feat_bf = (unsigned short*)alloc((size_t)NNODES * FDIM * 2);
    unsigned short* w2v_bf  = (unsigned short*)alloc((size_t)NNODES * LDIM * 2);
    unsigned short* Wvt_bf  = (unsigned short*)alloc((size_t)FDIM * FDIM * 2);   // [N=F][K=F]
    unsigned short* Wlt_bf  = (unsigned short*)alloc((size_t)LDIM * LDIM * 2);   // [N=L][K=L]
    unsigned short* W1t_bf  = (unsigned short*)alloc((size_t)HDDIM * DIN * 2);   // [N=HD][K=DIN]
    unsigned short* W2t_bf  = (unsigned short*)alloc((size_t)CDIM * HDDIM * 2);  // [N=C][K=HD]
    float* h_v   = (float*)alloc((size_t)NNODES * FDIM * 4);
    float* h_l   = (float*)alloc((size_t)NNODES * LDIM * 4);
    float* es_v  = (float*)alloc((size_t)NNODES * 4);
    float* ed_v  = (float*)alloc((size_t)NNODES * 4);
    float* es_l  = (float*)alloc((size_t)NNODES * 4);
    float* ed_l  = (float*)alloc((size_t)NNODES * 4);
    unsigned short* hv_new = (unsigned short*)alloc((size_t)NNODES * FDIM * 2);
    unsigned short* hl_new = (unsigned short*)alloc((size_t)NNODES * LDIM * 2);
    unsigned short* hdn_bf = (unsigned short*)alloc((size_t)NREAD * HDDIM * 2);
    unsigned short* spb    = (unsigned short*)alloc((size_t)NREAD * SDIM * 2);
    int* tab = (int*)alloc((size_t)NREAD * 2 * 4);

    auto cgrid = [](int n) { return dim3((n + 255) / 256); };

    // 1) conversions / transposes / gather tables
    cvt_f32_bf16<<<cgrid(NNODES * FDIM), 256, 0, stream>>>(feat, feat_bf, NNODES * FDIM);
    cvt_f32_bf16<<<cgrid(NNODES * LDIM), 256, 0, stream>>>(w2v, w2v_bf, NNODES * LDIM);
    cvt_transpose_bf16<<<cgrid(FDIM * FDIM), 256, 0, stream>>>(W_v, Wvt_bf, FDIM, FDIM);
    cvt_transpose_bf16<<<cgrid(LDIM * LDIM), 256, 0, stream>>>(W_l, Wlt_bf, LDIM, LDIM);
    cvt_transpose_bf16<<<cgrid(DIN * HDDIM), 256, 0, stream>>>(W1, W1t_bf, DIN, HDDIM);
    cvt_transpose_bf16<<<cgrid(HDDIM * CDIM), 256, 0, stream>>>(W2, W2t_bf, HDDIM, CDIM);
    build_tab<<<cgrid(NREAD), 256, 0, stream>>>(esrc, edst, reid, tab, NREAD);
    gather_spatial_bf16<<<cgrid(NREAD * SDIM), 256, 0, stream>>>(spat, reid, spb, NREAD * SDIM);

    // 2) h_v = feat @ W_v  (5120x1024x1024)
    {
        DirectLoader ldA{feat_bf, NNODES, FDIM, FDIM};
        dim3 g((FDIM + 127) / 128, (NNODES + 127) / 128);
        gemm_wmma<DirectLoader, 0><<<g, 256, 0, stream>>>(
            ldA, Wvt_bf, NNODES, FDIM, FDIM, nullptr, h_v, nullptr, FDIM);
    }
    // 3) h_l = w2v @ W_l  (5120x300x300)
    {
        DirectLoader ldA{w2v_bf, NNODES, LDIM, LDIM};
        dim3 g((LDIM + 127) / 128, (NNODES + 127) / 128);
        gemm_wmma<DirectLoader, 0><<<g, 256, 0, stream>>>(
            ldA, Wlt_bf, NNODES, LDIM, LDIM, nullptr, h_l, nullptr, LDIM);
    }

    // 4) attention scores + structured segment-softmax aggregation
    row_scores<<<dim3(NNODES / 8), 256, 0, stream>>>(h_v, a_v, es_v, ed_v, NNODES, FDIM);
    row_scores<<<dim3(NNODES / 8), 256, 0, stream>>>(h_l, a_l, es_l, ed_l, NNODES, LDIM);
    gat_aggregate<<<dim3(NNODES), 128, 0, stream>>>(feat, h_v, es_v, ed_v, hv_new, FDIM);
    gat_aggregate<<<dim3(NNODES), 128, 0, stream>>>(w2v, h_l, es_l, ed_l, hl_new, LDIM);

    // 5) hdn = relu(gathered-concat @ W1 + b1)   (19456x2664x1024, fused gather)
    {
        GatherLoader ldA{tab, hv_new, hl_new, spb, NREAD};
        dim3 g((HDDIM + 127) / 128, (NREAD + 127) / 128);
        gemm_wmma<GatherLoader, 1><<<g, 256, 0, stream>>>(
            ldA, W1t_bf, NREAD, HDDIM, DIN, b1, nullptr, hdn_bf, HDDIM);
    }
    // 6) logits = hdn @ W2 + b2   (19456x1024x117) -> d_out fp32
    {
        DirectLoader ldA{hdn_bf, NREAD, HDDIM, HDDIM};
        dim3 g((CDIM + 127) / 128, (NREAD + 127) / 128);
        gemm_wmma<DirectLoader, 2><<<g, 256, 0, stream>>>(
            ldA, W2t_bf, NREAD, CDIM, HDDIM, b2, out, nullptr, CDIM);
    }
}
